// GATModel_32289564131628
// MI455X (gfx1250) — matrix operand order repaired and checked by reading, not compiled, not run on hardware
//
#include <hip/hip_runtime.h>
#include <hip/hip_bf16.h>
#include <math.h>

// ---------------------------------------------------------------------------
// GATv2 model forward for MI455X (gfx1250, wave32, WMMA).
// The reference's target branch feeds only a dead store -> we compute the
// source branch + MLP head only.
//
// Node-feature GEMMs (x@Wl, x@Wr per layer) use v_wmma_f32_16x16x32_bf16
// with f32 accumulation; fragments use contiguous k order (k = 16*half + s)
// in both A and B, not the ISA table order. Tile staging is branch-free (index clamping
// instead of zero-guards) with global_load_b128 and packed LDS stores.
// Everything else (edge attention, segment softmax via ordered-uint
// atomicMax, scatter aggregation, GraphNorm/BN) is bandwidth-bound
// VMEM/atomic work.
//
// Param flattening assumption (JAX pytree, alphabetical keys):
//  d_in[0..7]  : x_s, edge_index_s, edge_attr_s, xs_batch, x_t, ei_t, ea_t, xt_batch
//  d_in[8..11] : bn1.b, bn1.g, bn2.b, bn2.g
//  d_in[12..29]: gn1..gn6 each (alpha, b, w)
//  d_in[30..35]: lin1.W, lin1.b, lin2.W, lin2.b, lin3.W, lin3.b
//  d_in[36..42]: s1 (We, Wl, Wr, att, bias, bl, br)
//  d_in[43..49]: s2        d_in[50..56]: s3
//  d_in[57..77]: t1..t3 (unused: dead code in reference)
// ---------------------------------------------------------------------------

#define FDIM 128
#define EDIM 9
#define GDIM 64

typedef __attribute__((ext_vector_type(16))) __bf16 v16bf;
typedef __attribute__((ext_vector_type(8)))  float  v8f;

__device__ __forceinline__ unsigned short f2bf(float x) {
  unsigned int u = __float_as_uint(x);
  unsigned int r = u + 0x7FFFu + ((u >> 16) & 1u);   // round-to-nearest-even
  return (unsigned short)(r >> 16);
}
__device__ __forceinline__ unsigned int pack2bf(float a, float b) {
  return (unsigned int)f2bf(a) | ((unsigned int)f2bf(b) << 16);
}

// monotonic float <-> uint encoding for atomicMax-based segment max
__device__ __forceinline__ unsigned int fenc(float x) {
  unsigned int u = __float_as_uint(x);
  return (u & 0x80000000u) ? ~u : (u | 0x80000000u);
}
__device__ __forceinline__ float fdec(unsigned int e) {
  unsigned int u = (e & 0x80000000u) ? (e ^ 0x80000000u) : ~e;
  return __uint_as_float(u);
}

// ---------------------------------------------------------------------------
// WMMA GEMM: C[M,Nd] = A[M,K] @ B[K,Nd] + bias[Nd]   (A,B fp32 -> bf16)
// block = 256 threads = 8 waves; block tile 64x64.
// Each wave computes a 16x32 C strip: two 16x16 WMMA tiles sharing one
// A fragment (back-to-back v_wmma).
// Out-of-range rows/cols are CLAMPED on load (not guarded): a garbage A row r
// only feeds C row r, a garbage B col n only feeds C col n, and those are
// masked at the store. This keeps the staging loop branch-free with b128 loads.
// ---------------------------------------------------------------------------
#define TM 64
#define TN 64
#define TK 32

__global__ void __launch_bounds__(256)
gemm_bias_wmma(const float* __restrict__ A, const float* __restrict__ B,
               const float* __restrict__ bias, float* __restrict__ C,
               int M, int K, int Nd)
{
  __shared__ unsigned short Alds[TM][TK + 2];   // +2 pad: row stride 68B, pairs stay 4B-aligned
  __shared__ unsigned short Blds[TN][TK + 2];   // stored transposed: [n][k]

  const int rowBase = blockIdx.x * TM;
  const int colBase = blockIdx.y * TN;
  const int tid  = threadIdx.x;
  const int lane = tid & 31;
  const int wave = tid >> 5;
  const int wm   = wave >> 1;     // 0..3 (M tile)
  const int wnp  = wave & 1;      // 0..1 (pair of N tiles)
  const int half = lane >> 4;     // 0/1
  const int l15  = lane & 15;

  const int c0 = colBase + wnp * 32 + l15;
  const int c1 = c0 + 16;
  const float bv0 = bias[min(c0, Nd - 1)];
  const float bv1 = bias[min(c1, Nd - 1)];
  v8f acc0 = { bv0, bv0, bv0, bv0, bv0, bv0, bv0, bv0 };
  v8f acc1 = { bv1, bv1, bv1, bv1, bv1, bv1, bv1, bv1 };

  for (int kt = 0; kt < K; kt += TK) {
    // A tile: 64x32 f32 -> bf16, 512 float4 chunks, 2 per thread, rows clamped
    #pragma unroll
    for (int i = 0; i < 2; ++i) {
      int ch = tid + i * 256;
      int mr = ch >> 3, k4 = (ch & 7) * 4;
      int gr = min(rowBase + mr, M - 1);
      const float* ap = A + (long)gr * K + kt + k4;
      float4 v = *(const float4*)ap;
      if (kt + TK < K) __builtin_prefetch(ap + TK, 0, 0);   // global_prefetch_b8
      *(unsigned int*)&Alds[mr][k4]     = pack2bf(v.x, v.y);
      *(unsigned int*)&Alds[mr][k4 + 2] = pack2bf(v.z, v.w);
    }
    // B tile: 32x64 f32 -> bf16 transposed into [n][k]; cols clamped
    #pragma unroll
    for (int i = 0; i < 2; ++i) {
      int ch = tid + i * 256;
      int kk = ch >> 4, n4 = (ch & 15) * 4;
      int gc = min(colBase + n4, Nd - 4);
      float4 v = *(const float4*)(B + (long)(kt + kk) * Nd + gc);
      Blds[n4 + 0][kk] = f2bf(v.x);
      Blds[n4 + 1][kk] = f2bf(v.y);
      Blds[n4 + 2][kk] = f2bf(v.z);
      Blds[n4 + 3][kk] = f2bf(v.w);
    }
    __syncthreads();

    // Per-lane fragment packing in contiguous k order, k = half*16 + s, (16-bit A 16x32, B 32x16)
    union { v16bf v; unsigned short s[16]; } fa, fb0, fb1;
    const int mr  = wm * 16 + l15;
    const int nc0 = wnp * 32 + l15;
    const int nc1 = nc0 + 16;
    #pragma unroll
    for (int p = 0; p < 8; ++p) {
      int ka = half * 16 + 2 * p;                       // A: K = half*16 + j
      fa.s[2 * p]      = Alds[mr][ka];
      fa.s[2 * p + 1]  = Alds[mr][ka + 1];
      int kb = half * 16 + 2 * p;                       // B: K = half*16 + j
      fb0.s[2 * p]     = Blds[nc0][kb];
      fb0.s[2 * p + 1] = Blds[nc0][kb + 1];
      fb1.s[2 * p]     = Blds[nc1][kb];
      fb1.s[2 * p + 1] = Blds[nc1][kb + 1];
    }
    acc0 = __builtin_amdgcn_wmma_f32_16x16x32_bf16(
        false, fa.v, false, fb0.v, (short)0, acc0, false, false);
    acc1 = __builtin_amdgcn_wmma_f32_16x16x32_bf16(
        false, fa.v, false, fb1.v, (short)0, acc1, false, false);
    __syncthreads();
  }

  #pragma unroll
  for (int r = 0; r < 8; ++r) {                   // D: row = half*8 + r
    int row = rowBase + wm * 16 + half * 8 + r;
    if (row < M) {
      if (c0 < Nd) C[(long)row * Nd + c0] = acc0[r];
      if (c1 < Nd) C[(long)row * Nd + c1] = acc1[r];
    }
  }
}

// ---------------------------------------------------------------------------
// Utility / elementwise kernels
// ---------------------------------------------------------------------------
__global__ void fill_f32(float* p, float v, long n) {
  long t = blockIdx.x * (long)blockDim.x + threadIdx.x;
  if (t < n) p[t] = v;
}
__global__ void fill_u32(unsigned int* p, unsigned int v, long n) {
  long t = blockIdx.x * (long)blockDim.x + threadIdx.x;
  if (t < n) p[t] = v;
}
__global__ void bias_bcast(float* out, const float* __restrict__ bias, int n, int d) {
  long t = blockIdx.x * (long)blockDim.x + threadIdx.x;
  if (t < (long)n * d) out[t] = bias[t % d];
}

// self-loop edge attr: deg + ea column-sum per destination
__global__ void deg_easum(const int* __restrict__ dst, const float* __restrict__ ea,
                          float* deg, float* easum, int E) {
  int e = blockIdx.x * blockDim.x + threadIdx.x;
  if (e >= E) return;
  int dd = dst[e];
  atomicAdd(deg + dd, 1.f);
  #pragma unroll
  for (int k = 0; k < EDIM; ++k)
    atomicAdd(easum + (long)dd * EDIM + k, ea[(long)e * EDIM + k]);
}
__global__ void ealoop_div(const float* __restrict__ easum, const float* __restrict__ deg,
                           float* ea_loop, int N) {
  long t = blockIdx.x * (long)blockDim.x + threadIdx.x;
  if (t >= (long)N * EDIM) return;
  ea_loop[t] = easum[t] / fmaxf(deg[t / EDIM], 1.f);
}

// attention logits per (edge, head): lrelu(xl[s]+xr[d]+ea@We, 0.2) . att[h]
__global__ void att_logits(const float* __restrict__ xl, const float* __restrict__ xr,
                           const int* __restrict__ src, const int* __restrict__ dst,
                           const float* __restrict__ ea, const float* __restrict__ ea_loop,
                           const float* __restrict__ We, const float* __restrict__ att,
                           float* logits, unsigned int* lmax,
                           int E, int N, int H, int C) {
  long t = blockIdx.x * (long)blockDim.x + threadIdx.x;
  if (t >= (long)(E + N) * H) return;
  int h = (int)(t % H);
  int e = (int)(t / H);
  int s  = (e < E) ? src[e] : (e - E);
  int dd = (e < E) ? dst[e] : (e - E);
  const float* eav = (e < E) ? (ea + (long)e * EDIM) : (ea_loop + (long)(e - E) * EDIM);
  float ev[EDIM];
  #pragma unroll
  for (int k = 0; k < EDIM; ++k) ev[k] = eav[k];
  int d = H * C;
  const float* xls  = xl + (long)s * d + h * C;
  const float* xrd  = xr + (long)dd * d + h * C;
  const float* atth = att + h * C;
  float acc = 0.f;
  for (int c = 0; c < C; ++c) {
    int j = h * C + c;
    float ee = 0.f;
    #pragma unroll
    for (int k = 0; k < EDIM; ++k) ee += ev[k] * We[k * d + j];
    float m = xls[c] + xrd[c] + ee;
    m = (m >= 0.f) ? m : 0.2f * m;
    acc += m * atth[c];
  }
  logits[t] = acc;
  atomicMax(lmax + (long)dd * H + h, fenc(acc));
}

// w = exp(logit - max[dst]); den[dst] += w
__global__ void softmax_wden(const float* __restrict__ logits,
                             const unsigned int* __restrict__ lmax,
                             const int* __restrict__ dst,
                             float* wexp, float* den, int E, int N, int H) {
  long t = blockIdx.x * (long)blockDim.x + threadIdx.x;
  if (t >= (long)(E + N) * H) return;
  int h = (int)(t % H);
  int e = (int)(t / H);
  int dd = (e < E) ? dst[e] : (e - E);
  float w = expf(logits[t] - fdec(lmax[(long)dd * H + h]));
  wexp[t] = w;
  atomicAdd(den + (long)dd * H + h, w);
}

// out[dst] += xl[src] * alpha   (out pre-initialized with GATv2 bias)
__global__ void aggregate(const float* __restrict__ xl, const float* __restrict__ wexp,
                          const float* __restrict__ den,
                          const int* __restrict__ src, const int* __restrict__ dst,
                          float* out, int E, int N, int H, int C) {
  long t = blockIdx.x * (long)blockDim.x + threadIdx.x;
  int d = H * C;
  if (t >= (long)(E + N) * d) return;
  int j = (int)(t % d);
  int e = (int)(t / d);
  int h = j / C;
  int s  = (e < E) ? src[e] : (e - E);
  int dd = (e < E) ? dst[e] : (e - E);
  float alpha = wexp[(long)e * H + h] / fmaxf(den[(long)dd * H + h], 1e-16f);
  atomicAdd(out + (long)dd * d + j, xl[(long)s * d + j] * alpha);
}

// per-column mean and mean-of-squares (block per column)
__global__ void colstats(const float* __restrict__ x, int n, int d,
                         float* mean, float* meansq) {
  int c = blockIdx.x;
  float s = 0.f, sq = 0.f;
  for (int r = threadIdx.x; r < n; r += blockDim.x) {
    float v = x[(long)r * d + c];
    s += v; sq += v * v;
  }
  __shared__ float ss[256], ssq[256];
  ss[threadIdx.x] = s; ssq[threadIdx.x] = sq;
  __syncthreads();
  for (int o = blockDim.x >> 1; o > 0; o >>= 1) {
    if (threadIdx.x < (unsigned)o) {
      ss[threadIdx.x]  += ss[threadIdx.x + o];
      ssq[threadIdx.x] += ssq[threadIdx.x + o];
    }
    __syncthreads();
  }
  if (threadIdx.x == 0) { mean[c] = ss[0] / n; meansq[c] = ssq[0] / n; }
}

// GraphNorm + leaky relu: xc = x - a*mu; var = E[xc^2] = msq - mu^2*a*(2-a)
__global__ void gnorm_lrelu(const float* __restrict__ x,
                            const float* __restrict__ mean, const float* __restrict__ meansq,
                            const float* __restrict__ alpha, const float* __restrict__ w,
                            const float* __restrict__ b,
                            float* y, int n, int d, float slope) {
  long t = blockIdx.x * (long)blockDim.x + threadIdx.x;
  if (t >= (long)n * d) return;
  int c = (int)(t % d);
  float mu = mean[c], a = alpha[c];
  float var = meansq[c] - mu * mu * a * (2.f - a);
  float v = (x[t] - a * mu) * rsqrtf(var + 1e-5f) * w[c] + b[c];
  y[t] = (v >= 0.f) ? v : slope * v;
}

// BatchNorm + leaky relu
__global__ void bn_lrelu(const float* __restrict__ x,
                         const float* __restrict__ mean, const float* __restrict__ meansq,
                         const float* __restrict__ g, const float* __restrict__ b,
                         float* y, int n, int d, float slope) {
  long t = blockIdx.x * (long)blockDim.x + threadIdx.x;
  if (t >= (long)n * d) return;
  int c = (int)(t % d);
  float mu = mean[c];
  float var = meansq[c] - mu * mu;
  float v = (x[t] - mu) * rsqrtf(var + 1e-5f) * g[c] + b[c];
  y[t] = (v >= 0.f) ? v : slope * v;
}

// graph mean pool
__global__ void pool_sum(const float* __restrict__ x, const int* __restrict__ batch,
                         float* pool, float* cnt, int n, int d) {
  long t = blockIdx.x * (long)blockDim.x + threadIdx.x;
  if (t >= (long)n * d) return;
  int r = (int)(t / d), c = (int)(t % d);
  int b = batch[r];
  atomicAdd(pool + (long)b * d + c, x[t]);
  if (c == 0) atomicAdd(cnt + b, 1.f);
}
__global__ void pool_div(float* pool, const float* __restrict__ cnt, int g, int d) {
  long t = blockIdx.x * (long)blockDim.x + threadIdx.x;
  if (t >= (long)g * d) return;
  pool[t] /= fmaxf(cnt[t / d], 1.f);
}

// small dense layer (head)
__global__ void lin_small(const float* __restrict__ x, const float* __restrict__ W,
                          const float* __restrict__ b, float* y, int n, int K, int O) {
  long t = blockIdx.x * (long)blockDim.x + threadIdx.x;
  if (t >= (long)n * O) return;
  int r = (int)(t / O), o = (int)(t % O);
  float acc = b[o];
  for (int k = 0; k < K; ++k) acc += x[(long)r * K + k] * W[(long)k * O + o];
  y[t] = acc;
}

// final layer: logits then sigmoid, concatenated in d_out
__global__ void final_out(const float* __restrict__ x, const float* __restrict__ W,
                          const float* __restrict__ b, float* out, int n, int K, int O) {
  long t = blockIdx.x * (long)blockDim.x + threadIdx.x;
  if (t >= (long)n * O) return;
  int r = (int)(t / O), o = (int)(t % O);
  float acc = b[o];
  for (int k = 0; k < K; ++k) acc += x[(long)r * K + k] * W[(long)k * O + o];
  out[t] = acc;
  out[(long)n * O + t] = 1.f / (1.f + expf(-acc));
}

// ---------------------------------------------------------------------------
static inline unsigned blks(long n) { return (unsigned)((n + 255) / 256); }

static void gat_layer(const float* xin, int cin, int H, int C,
                      const int* src, const int* dstp,
                      const float* ea, const float* ealp,
                      const float* We, const float* Wl, const float* Wr,
                      const float* att, const float* bias,
                      const float* bl, const float* br,
                      const float* gn_a, const float* gn_b, const float* gn_w,
                      float* xl, float* xr, float* aggb, float* outb,
                      float* logits, float* wexp, unsigned int* lmax, float* den,
                      float* meanb, float* msqb,
                      int N, int E, hipStream_t stream)
{
  const int d = H * C;
  dim3 gg((N + TM - 1) / TM, (d + TN - 1) / TN);
  gemm_bias_wmma<<<gg, 256, 0, stream>>>(xin, Wl, bl, xl, N, cin, d);
  gemm_bias_wmma<<<gg, 256, 0, stream>>>(xin, Wr, br, xr, N, cin, d);

  const long nh = (long)N * H;
  fill_u32<<<blks(nh), 256, 0, stream>>>(lmax, 0x007FFFFFu, nh);   // fenc(-inf)
  fill_f32<<<blks(nh), 256, 0, stream>>>(den, 0.f, nh);

  const long eh = (long)(E + N) * H;
  att_logits<<<blks(eh), 256, 0, stream>>>(xl, xr, src, dstp, ea, ealp, We, att,
                                           logits, lmax, E, N, H, C);
  softmax_wden<<<blks(eh), 256, 0, stream>>>(logits, lmax, dstp, wexp, den, E, N, H);

  const long nd = (long)N * d;
  bias_bcast<<<blks(nd), 256, 0, stream>>>(aggb, bias, N, d);
  const long ed = (long)(E + N) * d;
  aggregate<<<blks(ed), 256, 0, stream>>>(xl, wexp, den, src, dstp, aggb, E, N, H, C);

  colstats<<<d, 256, 0, stream>>>(aggb, N, d, meanb, msqb);
  gnorm_lrelu<<<blks(nd), 256, 0, stream>>>(aggb, meanb, msqb, gn_a, gn_w, gn_b,
                                            outb, N, d, 0.01f);
}

extern "C" void kernel_launch(void* const* d_in, const int* in_sizes, int n_in,
                              void* d_out, int out_size, void* d_ws, size_t ws_size,
                              hipStream_t stream)
{
  (void)n_in; (void)out_size; (void)ws_size;
  const float* x_s      = (const float*)d_in[0];
  const int*   ei       = (const int*)d_in[1];
  const float* ea       = (const float*)d_in[2];
  const int*   xs_batch = (const int*)d_in[3];
  const int N = in_sizes[0] / FDIM;
  const int E = in_sizes[1] / 2;
  const int* src  = ei;
  const int* dstp = ei + E;

  const float* P[78];
  for (int i = 8; i < 78; ++i) P[i] = (const float*)d_in[i];

  // workspace carve-out (f32 units)
  float* ws = (float*)d_ws;
  size_t off = 0;
  auto A = [&](long n) { float* p = ws + off; off += (size_t)n; return p; };
  float* xl    = A((long)N * 512);
  float* xr    = A((long)N * 512);
  float* buf0  = A((long)N * 512);      // aggregation buffer
  float* buf1  = A((long)N * 512);      // layer outputs (L1, L3)
  float* buf2  = A((long)N * 128);      // layer output (L2)
  float* logit = A((long)(E + N) * 8);
  float* wexp  = A((long)(E + N) * 8);
  unsigned int* lmax = (unsigned int*)A((long)N * 8);
  float* den   = A((long)N * 8);
  float* deg   = A(N);
  float* easum = A((long)N * EDIM);
  float* ealp  = A((long)N * EDIM);
  float* meanb = A(1317);
  float* msqb  = A(1317);
  float* pool  = A((long)GDIM * 16);
  float* cnt   = A(GDIM);
  float* h1    = A((long)GDIM * 329);
  float* h2    = A((long)GDIM * 658);

  // self-loop edge attributes: segment mean of ea over destinations
  fill_f32<<<blks(N), 256, 0, stream>>>(deg, 0.f, N);
  fill_f32<<<blks((long)N * EDIM), 256, 0, stream>>>(easum, 0.f, (long)N * EDIM);
  deg_easum<<<blks(E), 256, 0, stream>>>(dstp, ea, deg, easum, E);
  ealoop_div<<<blks((long)N * EDIM), 256, 0, stream>>>(easum, deg, ealp, N);

  // --- source branch (target branch is dead code in the reference) ---
  // s1: We,Wl,Wr,att,bias,bl,br = P[36..42]; gn1 alpha,b,w = P[12..14]
  gat_layer(x_s, FDIM, 8, 64, src, dstp, ea, ealp,
            P[36], P[37], P[38], P[39], P[40], P[41], P[42],
            P[12], P[13], P[14],
            xl, xr, buf0, buf1, logit, wexp, lmax, den, meanb, msqb, N, E, stream);
  // s2 = P[43..49]; gn2 = P[15..17]
  gat_layer(buf1, 512, 4, 32, src, dstp, ea, ealp,
            P[43], P[44], P[45], P[46], P[47], P[48], P[49],
            P[15], P[16], P[17],
            xl, xr, buf0, buf2, logit, wexp, lmax, den, meanb, msqb, N, E, stream);
  // s3 = P[50..56]; gn3 = P[18..20]
  gat_layer(buf2, 128, 1, 16, src, dstp, ea, ealp,
            P[50], P[51], P[52], P[53], P[54], P[55], P[56],
            P[18], P[19], P[20],
            xl, xr, buf0, buf1, logit, wexp, lmax, den, meanb, msqb, N, E, stream);

  // mean pool over graphs
  fill_f32<<<blks((long)GDIM * 16), 256, 0, stream>>>(pool, 0.f, (long)GDIM * 16);
  fill_f32<<<blks(GDIM), 256, 0, stream>>>(cnt, 0.f, GDIM);
  pool_sum<<<blks((long)N * 16), 256, 0, stream>>>(buf1, xs_batch, pool, cnt, N, 16);
  pool_div<<<blks((long)GDIM * 16), 256, 0, stream>>>(pool, cnt, GDIM, 16);

  // MLP head: lin1 -> bn1 -> lrelu -> lin2 -> bn2 -> lrelu -> lin3 -> sigmoid
  lin_small<<<blks((long)GDIM * 329), 256, 0, stream>>>(pool, P[30], P[31], h1, GDIM, 16, 329);
  colstats<<<329, 256, 0, stream>>>(h1, GDIM, 329, meanb, msqb);
  bn_lrelu<<<blks((long)GDIM * 329), 256, 0, stream>>>(h1, meanb, msqb, P[9], P[8],
                                                       h1, GDIM, 329, 0.01f);
  lin_small<<<blks((long)GDIM * 658), 256, 0, stream>>>(h1, P[32], P[33], h2, GDIM, 329, 658);
  colstats<<<658, 256, 0, stream>>>(h2, GDIM, 658, meanb, msqb);
  bn_lrelu<<<blks((long)GDIM * 658), 256, 0, stream>>>(h2, meanb, msqb, P[11], P[10],
                                                       h2, GDIM, 658, 0.01f);
  final_out<<<blks((long)GDIM * 1317), 256, 0, stream>>>(h2, P[34], P[35],
                                                         (float*)d_out, GDIM, 658, 1317);
}
